// VectorQuantizer_19069654794346
// MI455X (gfx1250) — matrix operand order repaired and checked by reading, not compiled, not run on hardware
//
#include <hip/hip_runtime.h>

#define NROWS  131072   // 512*256
#define DDIM   128
#define KCB    1024
#define BETA_C 0.25f

typedef __attribute__((ext_vector_type(16))) __bf16 v16bf;
typedef __attribute__((ext_vector_type(8)))  float  v8f;

union BF16Frag { v16bf v; uint4 q[2]; unsigned short u[16]; };

__device__ __forceinline__ unsigned short f2bf(float f) {
  unsigned int x = __float_as_uint(f);
  x += 0x7FFFu + ((x >> 16) & 1u);           // round-to-nearest-even
  return (unsigned short)(x >> 16);
}

// ---------------------------------------------------------------- prep:
// per-codebook-column squared L2 norm  (K = 1024 columns, D = 128)
__global__ __launch_bounds__(256) void vq_prep(const float* __restrict__ emb,
                                               float* __restrict__ enorm) {
  int k = blockIdx.x * 256 + threadIdx.x;
  if (k < KCB) {
    float s = 0.f;
    #pragma unroll 4
    for (int d = 0; d < DDIM; ++d) { float e = emb[(size_t)d * KCB + k]; s += e * e; }
    enorm[k] = s;
  }
}

// ---------------------------------------------------------------- argmin:
// block = 128 rows (8 waves x 16-row A tiles), full K sweep with bf16 WMMA.
// A holds -2*x, accumulator starts at ||e||^2, so dist falls out of the WMMA.
__global__ __launch_bounds__(256) void vq_argmin(const float* __restrict__ x,
                                                 const float* __restrict__ emb,
                                                 const float* __restrict__ enorm,
                                                 int* __restrict__ idxOut) {
  // x tile (scaled by -2), bf16, row stride 136 u16 = 272 B (16B aligned)
  __shared__ __align__(16) unsigned short xs[128][136];
  // codebook chunk staged fragment-major: [sub][step][lane][element]
  __shared__ __align__(16) unsigned short esf[4][4][32][16];
  __shared__ float ens[KCB];

  const int t    = threadIdx.x;
  const int lane = t & 31;
  const int wave = t >> 5;
  const int rowBlock = blockIdx.x * 128;

  for (int i = t; i < KCB; i += 256) ens[i] = enorm[i];

  // stage -2*x tile as bf16 (float4 global loads, 64-bit LDS stores)
  {
    const float4* x4 = (const float4*)x;
    for (int i = t; i < 128 * 32; i += 256) {
      int r = i >> 5, c4 = (i & 31) << 2;
      float4 v = x4[(size_t)(rowBlock + r) * 32 + (i & 31)];
      uint2 w;
      w.x = (unsigned int)f2bf(-2.0f * v.x) | ((unsigned int)f2bf(-2.0f * v.y) << 16);
      w.y = (unsigned int)f2bf(-2.0f * v.z) | ((unsigned int)f2bf(-2.0f * v.w) << 16);
      *(uint2*)&xs[r][c4] = w;
    }
  }
  __syncthreads();

  // A fragments: 16x32 bf16 (ISA 7.12.2). Lane's 16 elements are one
  // contiguous 16-element run (k = 16*g + e) -> 2 x ds_load_b128 each.
  const int m = lane & 15;
  const int g = lane >> 4;
  BF16Frag afr[4];
  #pragma unroll
  for (int s = 0; s < 4; ++s) {
    const unsigned short* rowp = &xs[wave * 16 + m][s * 32 + 16 * g];
    afr[s].q[0] = *(const uint4*)(rowp);
    afr[s].q[1] = *(const uint4*)(rowp + 8);
  }

  float best[8]; int bidx[8];
  #pragma unroll
  for (int j = 0; j < 8; ++j) { best[j] = 3.4e38f; bidx[j] = 0; }

  for (int kc = 0; kc < KCB; kc += 64) {
    __syncthreads();
    // stage 128x64 codebook chunk into fragment-major layout.
    // Work item w: rows d0,d0+1 (even d0) x 4 consecutive k ->
    // 2 x global_load_b128, then 4 packed ds_store_b32
    // (d maps to element e=d&15, so the (d0,d0+1) pair is one aligned dword)
    for (int w = t; w < 1024; w += 256) {
      int d0    = (w >> 4) << 1;
      int kbase = (w & 15) << 2;
      float4 a = *(const float4*)&emb[(size_t)d0 * KCB + kc + kbase];
      float4 b = *(const float4*)&emb[(size_t)(d0 + 1) * KCB + kc + kbase];
      int sub  = kbase >> 4;
      int stp  = d0 >> 5;
      int gb   = (d0 >> 4) & 1;
      int e    = d0 & 15;
      float av[4] = {a.x, a.y, a.z, a.w};
      float bv[4] = {b.x, b.y, b.z, b.w};
      #pragma unroll
      for (int j = 0; j < 4; ++j) {
        int k  = kbase + j;
        int ln = (k & 15) + 16 * gb;
        unsigned int p = (unsigned int)f2bf(av[j]) |
                         ((unsigned int)f2bf(bv[j]) << 16);
        *(unsigned int*)&esf[sub][stp][ln][e] = p;
      }
    }
    if (kc + 64 < KCB)   // hint next chunk -> global_prefetch_b8
      __builtin_prefetch(emb + (size_t)(t & 127) * KCB + kc + 64, 0, 1);
    __syncthreads();

    #pragma unroll
    for (int sub = 0; sub < 4; ++sub) {
      // fetch all four B step-fragments first (8 x ds_load_b128)
      BF16Frag bfr[4];
      #pragma unroll
      for (int s = 0; s < 4; ++s) {
        const uint4* p = (const uint4*)&esf[sub][s][lane][0];
        bfr[s].q[0] = p[0];
        bfr[s].q[1] = p[1];
      }

      int   kGlobal = kc + sub * 16 + m;
      float en      = ens[kGlobal];
      v8f c;
      #pragma unroll
      for (int j = 0; j < 8; ++j) c[j] = en;   // dist accumulates in-place

      #pragma unroll
      for (int s = 0; s < 4; ++s)
        c = __builtin_amdgcn_wmma_f32_16x16x32_bf16(
              false, afr[s].v, false, bfr[s].v, (short)0, c, false, false);

      // strict < keeps the first (lowest-k) minimum within this lane,
      // since kGlobal increases monotonically over the sweep
      #pragma unroll
      for (int j = 0; j < 8; ++j) {
        float dist = c[j];
        if (dist < best[j]) { best[j] = dist; bidx[j] = kGlobal; }
      }
    }
  }

  // argmin across the 16 lanes sharing each C row (wave32 butterfly);
  // index tie-break only needed here
  #pragma unroll
  for (int j = 0; j < 8; ++j) {
    float b  = best[j];
    int   bi = bidx[j];
    #pragma unroll
    for (int off = 8; off >= 1; off >>= 1) {
      float ob = __shfl_xor(b,  off, 32);
      int   oi = __shfl_xor(bi, off, 32);
      if (ob < b || (ob == b && oi < bi)) { b = ob; bi = oi; }
    }
    if (m == 0)
      idxOut[rowBlock + wave * 16 + 8 * g + j] = bi;
  }
}

// ---------------------------------------------------------------- gather + loss partials
// one float4 (4 elements of one row) per thread
__global__ __launch_bounds__(256) void vq_gather(const float* __restrict__ x,
                                                 const float* __restrict__ emb,
                                                 const int* __restrict__ idx,
                                                 float* __restrict__ out,
                                                 float* __restrict__ blockSums) {
  __shared__ float red[256];
  int t4  = blockIdx.x * 256 + threadIdx.x;   // float4 index
  int row = t4 >> 5;
  int d0  = (t4 & 31) << 2;
  int k   = idx[row];

  float4 xv = ((const float4*)x)[t4];
  float4 q;
  q.x = emb[(size_t)(d0 + 0) * KCB + k];
  q.y = emb[(size_t)(d0 + 1) * KCB + k];
  q.z = emb[(size_t)(d0 + 2) * KCB + k];
  q.w = emb[(size_t)(d0 + 3) * KCB + k];

  float4 o;
  o.x = xv.x + (q.x - xv.x);
  o.y = xv.y + (q.y - xv.y);
  o.z = xv.z + (q.z - xv.z);
  o.w = xv.w + (q.w - xv.w);
  ((float4*)out)[t4] = o;

  float dx = q.x - xv.x, dy = q.y - xv.y, dz = q.z - xv.z, dw = q.w - xv.w;
  red[threadIdx.x] = dx * dx + dy * dy + dz * dz + dw * dw;
  __syncthreads();
  for (int s = 128; s > 0; s >>= 1) {
    if (threadIdx.x < s) red[threadIdx.x] += red[threadIdx.x + s];
    __syncthreads();
  }
  if (threadIdx.x == 0) blockSums[blockIdx.x] = red[0];
}

// ---------------------------------------------------------------- finalize scalars
__global__ __launch_bounds__(256) void vq_finalize(const float* __restrict__ blockSums,
                                                   float* __restrict__ out) {
  __shared__ float red[256];
  float s = 0.f;
  for (int i = threadIdx.x; i < (NROWS * DDIM) / 1024; i += 256) s += blockSums[i];
  red[threadIdx.x] = s;
  __syncthreads();
  for (int k = 128; k > 0; k >>= 1) {
    if (threadIdx.x < k) red[threadIdx.x] += red[threadIdx.x + k];
    __syncthreads();
  }
  if (threadIdx.x == 0) {
    float mse = red[0] / (float)(NROWS * DDIM);
    out[(size_t)NROWS * DDIM]     = mse;            // codebook_loss
    out[(size_t)NROWS * DDIM + 1] = BETA_C * mse;   // commitment_loss
  }
}

extern "C" void kernel_launch(void* const* d_in, const int* in_sizes, int n_in,
                              void* d_out, int out_size, void* d_ws, size_t ws_size,
                              hipStream_t stream) {
  const float* x   = (const float*)d_in[0];
  const float* emb = (const float*)d_in[1];
  float* out = (float*)d_out;

  char*  ws        = (char*)d_ws;
  float* enorm     = (float*)ws;                        // 4 KB
  int*   idx       = (int*)(ws + 4096);                 // 512 KB
  float* blockSums = (float*)(ws + 4096 + 524288);      // 64 KB (16384 partials)

  vq_prep    <<<4,                     256, 0, stream>>>(emb, enorm);
  vq_argmin  <<<NROWS / 128,           256, 0, stream>>>(x, emb, enorm, idx);
  vq_gather  <<<(NROWS * DDIM) / 1024, 256, 0, stream>>>(x, emb, idx, out, blockSums);
  vq_finalize<<<1,                     256, 0, stream>>>(blockSums, out);
}
